// DBSCAN_3917010174423
// MI455X (gfx1250) — matrix operand order repaired and checked by reading, not compiled, not run on hardware
//
#include <hip/hip_runtime.h>

// ---------------------------------------------------------------------------
// DBSCAN for N=8192 points, D=256 fp32, eps=22 (eps^2=484), min_samples=5.
// Hot loop: X @ X^T via v_wmma_f32_16x16x32_f16 with split-precision fp16
// (x = hi + lo; dot ~= hi*hi + hi*lo + lo*hi) for ~fp32 accuracy.
// 64x64 macro-tiles staged in LDS (132 KB) for 4x data-reuse; adjacency is
// an 8192x8192 bitmask (8 MB, L2-resident).
// ---------------------------------------------------------------------------

#define NPTS 8192
#define NDIM 256
#define EPS2 484.0f
#define MIN_SAMPLES 5
#define BIGC NPTS
#define NITER 8
#define LDSS 264   // padded row stride (elements): 528B -> 4-bank rotation/row

typedef __attribute__((ext_vector_type(16))) _Float16 v16h;
typedef __attribute__((ext_vector_type(8)))  _Float16 v8h;
typedef __attribute__((ext_vector_type(8)))  float    v8f;

static __device__ __forceinline__ unsigned wave_ballot32(bool p) {
#if __has_builtin(__builtin_amdgcn_ballot_w32)
    return __builtin_amdgcn_ballot_w32(p);
#else
    return (unsigned)__ballot((int)p);
#endif
}

static __device__ __forceinline__ int wave_min_i32(int v) {
    #pragma unroll
    for (int off = 16; off > 0; off >>= 1) {
        int o = __shfl_xor(v, off, 32);
        v = (o < v) ? o : v;
    }
    return v;
}

static __device__ __forceinline__ float wave_sum_f32(float v) {
    #pragma unroll
    for (int off = 16; off > 0; off >>= 1) v += __shfl_xor(v, off, 32);
    return v;
}

static __device__ __forceinline__ int wave_sum_i32(int v) {
    #pragma unroll
    for (int off = 16; off > 0; off >>= 1) v += __shfl_xor(v, off, 32);
    return v;
}

#define WMMA_F16(A, B, C) \
    __builtin_amdgcn_wmma_f32_16x16x32_f16(false, (A), false, (B), (short)0, (C), false, false)

// --- 1. fp32 -> (hi,lo) fp16 split ----------------------------------------
__global__ __launch_bounds__(256)
void dbscan_convert(const float* __restrict__ X,
                    _Float16* __restrict__ Xh, _Float16* __restrict__ Xl) {
    int i = blockIdx.x * 256 + threadIdx.x;   // over NPTS*NDIM elements
    float x = X[i];
    _Float16 h = (_Float16)x;
    Xh[i] = h;
    Xl[i] = (_Float16)(x - (float)h);
}

// --- 2. per-row squared norms (wave per row) ------------------------------
__global__ __launch_bounds__(256)
void dbscan_norms(const float* __restrict__ X, float* __restrict__ sqn) {
    int lane = threadIdx.x & 31;
    int row  = blockIdx.x * 8 + (threadIdx.x >> 5);
    const float* xr = X + (size_t)row * NDIM;
    float s = 0.f;
    #pragma unroll
    for (int k = lane; k < NDIM; k += 32) { float v = xr[k]; s += v * v; }
    s = wave_sum_f32(s);
    if (lane == 0) sqn[row] = s;
}

// --- 3. WMMA Gram tiles + adjacency bit-pack ------------------------------
// Block (8 waves) computes a 64x64 output macro-tile. A-rows and B-rows
// (hi+lo fp16, full K=256) are staged cooperatively in LDS, then each wave
// computes two 16x16 tiles sharing one A fragment: 6 WMMAs per K-chunk.
__global__ __launch_bounds__(256)
void dbscan_gemm_adj(const _Float16* __restrict__ Xh,
                     const _Float16* __restrict__ Xl,
                     const float* __restrict__ sqn,
                     unsigned short* __restrict__ adj16) {
    __shared__ _Float16 lds[4 * 64 * LDSS];           // 132 KB
    _Float16* ldsAh = lds;
    _Float16* ldsAl = lds + 1 * 64 * LDSS;
    _Float16* ldsBh = lds + 2 * 64 * LDSS;
    _Float16* ldsBl = lds + 3 * 64 * LDSS;

    const int t  = threadIdx.x;
    const int i0 = blockIdx.y * 64;
    const int j0 = blockIdx.x * 64;

    // Cooperative stage: 64 A-rows + 64 B-rows, hi+lo, 16B chunks.
    #pragma unroll
    for (int q = 0; q < 8; ++q) {
        int c   = t + 256 * q;        // 0..2047
        int row = c >> 5;             // 0..63
        int ko  = (c & 31) * 8;       // element offset within row
        *(v8h*)(ldsAh + row * LDSS + ko) = *(const v8h*)(Xh + (size_t)(i0 + row) * NDIM + ko);
        *(v8h*)(ldsAl + row * LDSS + ko) = *(const v8h*)(Xl + (size_t)(i0 + row) * NDIM + ko);
        *(v8h*)(ldsBh + row * LDSS + ko) = *(const v8h*)(Xh + (size_t)(j0 + row) * NDIM + ko);
        *(v8h*)(ldsBl + row * LDSS + ko) = *(const v8h*)(Xl + (size_t)(j0 + row) * NDIM + ko);
    }
    __syncthreads();

    const int lane = t & 31;
    const int wave = t >> 5;
    const int ti = wave >> 1;                 // 0..3  (i sub-tile)
    const int tj = wave & 1;                  // tiles tj and tj+2 (j sub-tiles)

    // A fragment (16x32 f16): lane&15 = M, lane>>4 selects K-half
    const int mA = lane & 15, kbaseA = (lane >> 4) * 16;
    // B fragment (32x16 f16): lane&15 = N, lane>>4 selects K-half
    const int nB = lane & 15, koffB = (lane >> 4) * 16;

    const _Float16* aRowH = ldsAh + (16 * ti + mA) * LDSS;
    const _Float16* aRowL = ldsAl + (16 * ti + mA) * LDSS;
    const _Float16* b0H   = ldsBh + (16 * tj + nB) * LDSS;
    const _Float16* b0L   = ldsBl + (16 * tj + nB) * LDSS;
    const _Float16* b1H   = ldsBh + (16 * (tj + 2) + nB) * LDSS;
    const _Float16* b1L   = ldsBl + (16 * (tj + 2) + nB) * LDSS;

    v8f c0 = {}, c1 = {};
    #pragma unroll
    for (int k0 = 0; k0 < NDIM; k0 += 32) {
        v8h a0h = *(const v8h*)(aRowH + k0 + kbaseA);
        v8h a1h = *(const v8h*)(aRowH + k0 + 8 + kbaseA);
        v8h a0l = *(const v8h*)(aRowL + k0 + kbaseA);
        v8h a1l = *(const v8h*)(aRowL + k0 + 8 + kbaseA);
        v16h ah = __builtin_shufflevector(a0h, a1h, 0,1,2,3,4,5,6,7,8,9,10,11,12,13,14,15);
        v16h al = __builtin_shufflevector(a0l, a1l, 0,1,2,3,4,5,6,7,8,9,10,11,12,13,14,15);

        v8h p0, p1;
        p0 = *(const v8h*)(b0H + k0 + koffB);
        p1 = *(const v8h*)(b0H + k0 + koffB + 8);
        v16h bh0 = __builtin_shufflevector(p0, p1, 0,1,2,3,4,5,6,7,8,9,10,11,12,13,14,15);
        p0 = *(const v8h*)(b0L + k0 + koffB);
        p1 = *(const v8h*)(b0L + k0 + koffB + 8);
        v16h bl0 = __builtin_shufflevector(p0, p1, 0,1,2,3,4,5,6,7,8,9,10,11,12,13,14,15);
        p0 = *(const v8h*)(b1H + k0 + koffB);
        p1 = *(const v8h*)(b1H + k0 + koffB + 8);
        v16h bh1 = __builtin_shufflevector(p0, p1, 0,1,2,3,4,5,6,7,8,9,10,11,12,13,14,15);
        p0 = *(const v8h*)(b1L + k0 + koffB);
        p1 = *(const v8h*)(b1L + k0 + koffB + 8);
        v16h bl1 = __builtin_shufflevector(p0, p1, 0,1,2,3,4,5,6,7,8,9,10,11,12,13,14,15);

        c0 = WMMA_F16(ah, bh0, c0);
        c0 = WMMA_F16(ah, bl0, c0);
        c0 = WMMA_F16(al, bh0, c0);
        c1 = WMMA_F16(ah, bh1, c1);
        c1 = WMMA_F16(ah, bl1, c1);
        c1 = WMMA_F16(al, bh1, c1);
    }

    // Epilogue: d2 = |xi|^2 + |xj|^2 - 2*dot <= eps^2, pack bits via ballot.
    // C/D layout: VGPR r -> lanes 0-15: M=r, lanes 16-31: M=r+8; N = lane&15.
    const int   jA = j0 + 16 * tj;
    const int   jB = j0 + 16 * (tj + 2);
    const float sqJ0 = sqn[jA + nB];
    const float sqJ1 = sqn[jB + nB];
    const int   rowsel = (lane < 16) ? 0 : 8;
    const int   iBase  = i0 + 16 * ti;
    #pragma unroll
    for (int r = 0; r < 8; ++r) {
        float sqI = sqn[iBase + r + rowsel];
        unsigned m0 = wave_ballot32(sqI + sqJ0 - 2.0f * c0[r] <= EPS2);
        unsigned m1 = wave_ballot32(sqI + sqJ1 - 2.0f * c1[r] <= EPS2);
        if (lane == 0) {
            size_t b0 = (size_t)(iBase + r)     * (NPTS / 16);
            size_t b8 = (size_t)(iBase + r + 8) * (NPTS / 16);
            adj16[b0 + (jA >> 4)] = (unsigned short)(m0 & 0xFFFFu);
            adj16[b0 + (jB >> 4)] = (unsigned short)(m1 & 0xFFFFu);
            adj16[b8 + (jA >> 4)] = (unsigned short)(m0 >> 16);
            adj16[b8 + (jB >> 4)] = (unsigned short)(m1 >> 16);
        }
    }
}

// --- 4. core detection (wave per row: popcount of adjacency row) ----------
__global__ __launch_bounds__(256)
void dbscan_core(const unsigned* __restrict__ adjw, unsigned* __restrict__ coreflag) {
    int lane = threadIdx.x & 31;
    int row  = blockIdx.x * 8 + (threadIdx.x >> 5);
    const unsigned* r = adjw + (size_t)row * (NPTS / 32);
    int cnt = 0;
    #pragma unroll 4
    for (int w = lane; w < NPTS / 32; w += 32) cnt += __popc(r[w]);
    cnt = wave_sum_i32(cnt);
    if (lane == 0) coreflag[row] = (cnt >= MIN_SAMPLES) ? 1u : 0u;
}

// --- 5. pack core flags into a bit vector (no atomics, deterministic) -----
__global__ void dbscan_corebits(const unsigned* __restrict__ coreflag,
                                unsigned* __restrict__ corebits) {
    int w = threadIdx.x;  // 256 words
    unsigned bits = 0;
    #pragma unroll
    for (int b = 0; b < 32; ++b) bits |= (coreflag[w * 32 + b] & 1u) << b;
    corebits[w] = bits;
}

// --- 6. comp init: comp0 = core ? idx : BIG -------------------------------
__global__ void dbscan_compinit(const unsigned* __restrict__ coreflag,
                                int* __restrict__ compA, int* __restrict__ compB) {
    int i = blockIdx.x * 256 + threadIdx.x;
    int v = coreflag[i] ? i : BIGC;
    compA[i] = v;
    compB[i] = v;
}

// --- 7. one min-label propagation pass (wave per row) ---------------------
__global__ __launch_bounds__(256)
void dbscan_prop(const unsigned* __restrict__ adjw,
                 const unsigned* __restrict__ corebits,
                 const unsigned* __restrict__ coreflag,
                 const int* __restrict__ compIn, int* __restrict__ compOut) {
    int lane = threadIdx.x & 31;
    int row  = blockIdx.x * 8 + (threadIdx.x >> 5);
    if (!coreflag[row]) { if (lane == 0) compOut[row] = BIGC; return; }
    const unsigned* r = adjw + (size_t)row * (NPTS / 32);
    int best = BIGC;
    for (int w = 0; w < NPTS / 32; ++w) {
        unsigned m = r[w] & corebits[w];
        if (m & (1u << lane)) {
            int v = compIn[w * 32 + lane];   // coalesced gather
            best = (v < best) ? v : best;
        }
    }
    best = wave_min_i32(best);
    int self = compIn[row];
    best = (self < best) ? self : best;
    if (lane == 0) compOut[row] = best;
}

// --- 8. pointer-jumping pass (double-buffered, deterministic) -------------
__global__ void dbscan_jump(const int* __restrict__ compIn, int* __restrict__ compOut) {
    int i = blockIdx.x * 256 + threadIdx.x;
    int c = compIn[i];
    if (c < BIGC) {
        int cc = compIn[c];
        c = (cc < c) ? cc : c;
    }
    compOut[i] = c;
}

// --- 9. number clusters: cid[i] = cumsum(is_rep)[i] - 1 (one block) -------
__global__ __launch_bounds__(1024)
void dbscan_scan(const unsigned* __restrict__ coreflag,
                 const int* __restrict__ comp, int* __restrict__ cidAt) {
    __shared__ int sums[1024];
    int t = threadIdx.x;
    int base = t * 8;
    int local[8];
    int s = 0;
    #pragma unroll
    for (int u = 0; u < 8; ++u) {
        int i = base + u;
        int isrep = (coreflag[i] && comp[i] == i) ? 1 : 0;
        local[u] = isrep;
        s += isrep;
    }
    sums[t] = s;
    __syncthreads();
    for (int off = 1; off < 1024; off <<= 1) {
        int v = sums[t];
        int add = (t >= off) ? sums[t - off] : 0;
        __syncthreads();
        sums[t] = v + add;
        __syncthreads();
    }
    int run = (t == 0) ? 0 : sums[t - 1];
    #pragma unroll
    for (int u = 0; u < 8; ++u) {
        run += local[u];
        cidAt[base + u] = run - 1;   // inclusive cumsum minus 1
    }
}

// --- 10. final labels (core + border via min-index core neighbor) ---------
__global__ void dbscan_label(const unsigned* __restrict__ coreflag,
                             const int* __restrict__ comp,
                             const int* __restrict__ cidAt,
                             const unsigned* __restrict__ adjw,
                             const unsigned* __restrict__ corebits,
                             float* __restrict__ out) {
    int i = blockIdx.x * 256 + threadIdx.x;
    int label = -1;
    if (coreflag[i]) {
        label = cidAt[comp[i]];
    } else {
        const unsigned* r = adjw + (size_t)i * (NPTS / 32);
        for (int w = 0; w < NPTS / 32; ++w) {
            unsigned m = r[w] & corebits[w];
            if (m) { int j = w * 32 + (__ffs(m) - 1); label = cidAt[comp[j]]; break; }
        }
    }
    out[i] = (float)label;
}

// ---------------------------------------------------------------------------
extern "C" void kernel_launch(void* const* d_in, const int* in_sizes, int n_in,
                              void* d_out, int out_size, void* d_ws, size_t ws_size,
                              hipStream_t stream) {
    (void)in_sizes; (void)n_in; (void)out_size; (void)ws_size;
    const float* X = (const float*)d_in[0];
    float* out = (float*)d_out;

    char* ws = (char*)d_ws;
    size_t off = 0;
    _Float16* Xh  = (_Float16*)(ws + off); off += (size_t)NPTS * NDIM * 2;   // 4 MB
    _Float16* Xl  = (_Float16*)(ws + off); off += (size_t)NPTS * NDIM * 2;   // 4 MB
    unsigned short* adj16 = (unsigned short*)(ws + off);
    unsigned*       adjw  = (unsigned*)(ws + off); off += (size_t)NPTS * (NPTS / 8); // 8 MB
    float*    sqn      = (float*)(ws + off);    off += NPTS * 4;
    unsigned* coreflag = (unsigned*)(ws + off); off += NPTS * 4;
    int*      compA    = (int*)(ws + off);      off += NPTS * 4;
    int*      compB    = (int*)(ws + off);      off += NPTS * 4;
    int*      cidAt    = (int*)(ws + off);      off += NPTS * 4;
    unsigned* corebits = (unsigned*)(ws + off); off += (NPTS / 32) * 4;

    // 1-2: fp16 split + norms
    dbscan_convert<<<(NPTS * NDIM) / 256, 256, 0, stream>>>(X, Xh, Xl);
    dbscan_norms<<<NPTS / 8, 256, 0, stream>>>(X, sqn);

    // 3: WMMA Gram + adjacency bits (64x64 macro-tiles, LDS-staged)
    dbscan_gemm_adj<<<dim3(NPTS / 64, NPTS / 64, 1), 256, 0, stream>>>(Xh, Xl, sqn, adj16);

    // 4-6: core points, core bitvector, comp init
    dbscan_core<<<NPTS / 8, 256, 0, stream>>>(adjw, coreflag);
    dbscan_corebits<<<1, NPTS / 32, 0, stream>>>(coreflag, corebits);
    dbscan_compinit<<<NPTS / 256, 256, 0, stream>>>(coreflag, compA, compB);

    // 7-8: fixed-count label propagation + pointer jumping (deterministic)
    for (int it = 0; it < NITER; ++it) {
        dbscan_prop<<<NPTS / 8, 256, 0, stream>>>(adjw, corebits, coreflag, compA, compB);
        dbscan_jump<<<NPTS / 256, 256, 0, stream>>>(compB, compA);
    }

    // 9-10: cluster numbering + final labels
    dbscan_scan<<<1, 1024, 0, stream>>>(coreflag, compA, cidAt);
    dbscan_label<<<NPTS / 256, 256, 0, stream>>>(coreflag, compA, cidAt, adjw, corebits, out);
}
